// TemporalEncoder_46557445488651
// MI455X (gfx1250) — compile-verified, matrix operand order checked by reading, not run on hardware
//
#include <hip/hip_runtime.h>
#include <hip/hip_bf16.h>
#include <math.h>

typedef __attribute__((ext_vector_type(16))) _Float16 v16h;
typedef __attribute__((ext_vector_type(8)))  _Float16 v8h;
typedef __attribute__((ext_vector_type(8)))  float    v8f;
typedef __attribute__((ext_vector_type(4)))  float    v4f;

#define HID        64
#define T_STEPS    50
#define SEQ_PER_WG 16

union V16H { v16h v; v8h h[2]; };

__device__ __forceinline__ float fast_tanh(float x) {
#if __has_builtin(__builtin_amdgcn_tanhf)
  return __builtin_amdgcn_tanhf(x);
#else
  return tanhf(x);
#endif
}
__device__ __forceinline__ float fast_sigmoid(float x) {
  // sigmoid(x) = 0.5*tanh(0.5*x) + 0.5
  return fmaf(0.5f, fast_tanh(0.5f * x), 0.5f);
}

// Grid: one block of 128 threads (4 waves) per 16 sequences.
// Wave w owns hidden slice [16w, 16w+16) and its 4 gate tiles (i,f,g,o).
__global__ __launch_bounds__(128) void lstm_wmma_kernel(
    const float* __restrict__ dynamic,   // [BN, 50]
    const float* __restrict__ W_ih,      // [256, 1]
    const float* __restrict__ W_hh,      // [256, 64] row-major
    const float* __restrict__ b_ih,      // [256]
    const float* __restrict__ b_hh,      // [256]
    float* __restrict__ out)             // [BN, 64]
{
  __shared__ __align__(16) _Float16 hbuf[2][SEQ_PER_WG][HID];

  const int tid   = threadIdx.x;
  const int wid   = tid >> 5;      // wave id 0..3 = hidden-slice id
  const int lane  = tid & 31;
  const int m     = lane & 15;     // tile row (A) / sequence (B,C,D lanes)
  const int hi    = lane >> 4;     // lane half
  const int hioff = hi * 8;

  const long seq = (long)blockIdx.x * SEQ_PER_WG + m;

  // ---- W_hh A-fragments (f16, 16x32 A-layout), 4 gate tiles x 2 K-blocks.
  // A layout (16-bit, 16x32): lanes 0-15 M=0..15 hold K pairs
  // {0,1},{2,3},{4,5},{6,7},{16,17},{18,19},{20,21},{22,23};
  // lanes 16-31 hold the same rows with K+8.
  v16h afrag[4][2];
  for (int gt = 0; gt < 4; ++gt) {
    const int gate = (wid + 4 * gt) * 16 + m;
    const float* row = W_hh + gate * HID;
    for (int kb = 0; kb < 2; ++kb) {
      const float* p = row + kb * 32 + hioff;
      v16h f;
      #pragma unroll
      for (int e = 0; e < 8; ++e) f[e]     = (_Float16)p[e];
      #pragma unroll
      for (int e = 0; e < 8; ++e) f[8 + e] = (_Float16)p[16 + e];
      afrag[gt][kb] = f;
    }
  }

  // ---- Resident bias and w_in in the C/D layout: VGPR r, this lane-half
  // corresponds to gate index = tile*16 + hioff + r.
  v8f bias_v[4], win_v[4];
  for (int gt = 0; gt < 4; ++gt) {
    #pragma unroll
    for (int r = 0; r < 8; ++r) {
      const int gate = (wid + 4 * gt) * 16 + hioff + r;
      bias_v[gt][r] = b_ih[gate] + b_hh[gate];
      win_v[gt][r]  = W_ih[gate];
    }
  }

  // Zero h buffer 0 (h0 = 0).
  {
    _Float16* z = &hbuf[0][0][0];
    for (int i = tid; i < SEQ_PER_WG * HID; i += 128) z[i] = (_Float16)0.0f;
  }

  const float* xp = dynamic + seq * T_STEPS;

  v8f c_st = {};   // cell state, f32, resident all 50 steps
  v8f h_st = {};   // last h (f32) for final output
  int cur = 0;

  for (int t = 0; t < T_STEPS; ++t) {
    __syncthreads();   // previous step's h stores visible; buffer reusable

    const float x = xp[t];

    // ---- Build B fragments (h^T, K=hidden, N=seq) from LDS, lane-local:
    // lane (N=m, half hi), K-block kb: 8 halfs at hrow[kb*32+hioff],
    // 8 halfs at hrow[kb*32+hioff+16]  -> two ds_load_b128.
    V16H bfrag[2];
    const _Float16* hrow = &hbuf[cur][m][0];
    #pragma unroll
    for (int kb = 0; kb < 2; ++kb) {
      bfrag[kb].h[0] = *(const v8h*)(hrow + kb * 32 + hioff);
      bfrag[kb].h[1] = *(const v8h*)(hrow + kb * 32 + hioff + 16);
    }

    // ---- gates = bias + x*w_in + W_hh x h   (f32 accumulate)
    v8f acc[4];
    #pragma unroll
    for (int gt = 0; gt < 4; ++gt) {
      v8f g0;
      #pragma unroll
      for (int r = 0; r < 8; ++r) g0[r] = fmaf(x, win_v[gt][r], bias_v[gt][r]);
      g0 = __builtin_amdgcn_wmma_f32_16x16x32_f16(
               false, afrag[gt][0], false, bfrag[0].v, (short)0, g0, false, false);
      g0 = __builtin_amdgcn_wmma_f32_16x16x32_f16(
               false, afrag[gt][1], false, bfrag[1].v, (short)0, g0, false, false);
      acc[gt] = g0;
    }

    // ---- LSTM cell (per-lane VALU + v_tanh)
    #pragma unroll
    for (int r = 0; r < 8; ++r) {
      const float iv = fast_sigmoid(acc[0][r]);
      const float fv = fast_sigmoid(acc[1][r]);
      const float gv = fast_tanh(acc[2][r]);
      const float ov = fast_sigmoid(acc[3][r]);
      const float cn = fmaf(fv, c_st[r], iv * gv);
      c_st[r] = cn;
      h_st[r] = ov * fast_tanh(cn);
    }

    // ---- Store h slice (f16) for next step: hidden j = 16*wid + hioff + r,
    // contiguous in r -> one ds_store_b128.
    v8h hh;
    #pragma unroll
    for (int r = 0; r < 8; ++r) hh[r] = (_Float16)h_st[r];
    *(v8h*)(&hbuf[cur ^ 1][m][wid * 16 + hioff]) = hh;

    cur ^= 1;
  }

  // ---- Final h (f32) to out[seq][16*wid + hioff + r]
  float* op = out + seq * HID + wid * 16 + hioff;
  v4f o0, o1;
  #pragma unroll
  for (int r = 0; r < 4; ++r) { o0[r] = h_st[r]; o1[r] = h_st[4 + r]; }
  *(v4f*)op       = o0;
  *((v4f*)op + 1) = o1;
}

extern "C" void kernel_launch(void* const* d_in, const int* in_sizes, int n_in,
                              void* d_out, int out_size, void* d_ws, size_t ws_size,
                              hipStream_t stream) {
  const float* dynamic = (const float*)d_in[0];   // [B,N,T]
  const float* W_ih    = (const float*)d_in[1];   // [4H,1]
  const float* W_hh    = (const float*)d_in[2];   // [4H,H]
  const float* b_ih    = (const float*)d_in[3];   // [4H]
  const float* b_hh    = (const float*)d_in[4];   // [4H]
  float* out = (float*)d_out;

  const int n_seq  = in_sizes[0] / T_STEPS;       // B*N = 262144
  const int blocks = n_seq / SEQ_PER_WG;          // 16384

  lstm_wmma_kernel<<<blocks, 128, 0, stream>>>(dynamic, W_ih, W_hh, b_ih, b_hh, out);
}
